// HSI_Attention_70368744178074
// MI455X (gfx1250) — matrix operand order repaired and checked by reading, not compiled, not run on hardware
//
#include <hip/hip_runtime.h>
#include <hip/hip_bf16.h>

// ---------- types ----------
typedef __attribute__((ext_vector_type(16))) __bf16        v16bf;
typedef __attribute__((ext_vector_type(8)))  float         v8f;
typedef __attribute__((ext_vector_type(4)))  unsigned int  u32x4;
typedef __attribute__((ext_vector_type(8)))  int           i32x8;
typedef __attribute__((ext_vector_type(4)))  int           i32x4;
typedef __attribute__((ext_vector_type(4)))  unsigned int  uv4;   // 16B chunk

union Frag16 { v16bf v; uv4 q[2]; };

#define NWIN   8192
#define NH     6
#define HD     32
#define C_IN   192
#define C2     384
#define NTOK   49        // 7*7
#define MPAD   64        // padded token count (4 tiles of 16)
#define KSTR   200       // bf16 row stride for xw/q/k LDS tiles (16B aligned)
#define SSTR   52        // score row stride (floats)
#define QSCALE 0.17677669529663687f   // 32^-0.5

__device__ __forceinline__ unsigned short f2bf(float f) {
    unsigned u = __float_as_uint(f);
    u += 0x7FFFu + ((u >> 16) & 1u);          // round-to-nearest-even
    return (unsigned short)(u >> 16);
}

// ---------- shared memory ----------
struct __align__(16) SMem {
    union U {
        float xwf[C_IN * NTOK];               // TDM staging: [c][m] f32  (37,632 B)
        float scores[NH * NTOK * SSTR];       // attn logits f32          (61,152 B)
    } u;
    unsigned short xwbf[MPAD * KSTR];         // [m][c]  bf16 (25,600 B)
    unsigned short qbf [MPAD * KSTR];         // [m][c]  bf16 (scaled q)
    unsigned short kbf [MPAD * KSTR];         // [m][c]  bf16
    float rowmax[NH * NTOK];
    float rowsuminv[NH * NTOK];
};                                            // total ~140,304 B

// ---------- kernel 0: W_qk f32 -> bf16 into d_ws ----------
__global__ void wcvt_kernel(const float* __restrict__ w, unsigned short* __restrict__ o, int n) {
    int i = blockIdx.x * 256 + threadIdx.x;
    if (i < n) o[i] = f2bf(w[i]);
}

// ---------- kernel 1: fused window-attention ----------
__global__ __launch_bounds__(256) void
win_attn_kernel(const float* __restrict__ x,
                const unsigned short* __restrict__ Wbf,   // [384][192] bf16
                const float* __restrict__ bqk,            // [384]
                float* __restrict__ out)                  // [8192][6][49][49]
{
    __shared__ SMem sm;

    const int win = blockIdx.x;
    const int b   = win >> 10;          // 32*32 windows per image
    const int rem = win & 1023;
    const int hr  = rem >> 5;
    const int wr  = rem & 31;

    const int tid  = threadIdx.x;
    const int wid  = tid >> 5;
    const int lane = tid & 31;
    const int ln   = lane & 15;
    const int lh   = lane >> 4;

    // ---- phase 0: TDM load of the 7x7x192 f32 window tile into LDS ----
    if (wid == 0) {
        unsigned long long gaddr = (unsigned long long)(size_t)x +
            4ull * (((unsigned long long)b * C_IN * 224ull + (unsigned long long)(hr * 7)) * 224ull
                    + (unsigned long long)(wr * 7));
        unsigned lds_off = (unsigned)(size_t)(sm.u.xwf);

        u32x4 g0 = { 1u,                                   // count=1 valid descriptor
                     lds_off,
                     (unsigned)gaddr,
                     (unsigned)(gaddr >> 32) | 0x80000000u }; // type=2 ("image")
        i32x8 g1 = { (int)0x00020000,                      // data_size = 4B
                     (int)(224u << 16),                    // tensor_dim0 = 224 (lo16)
                     (int)(224u << 16),                    // tensor_dim1 = 224 (lo16)
                     (int)(7u << 16),                      // tile_dim0 = 7
                     (int)(7u | (192u << 16)),             // tile_dim1 = 7, tile_dim2 = 192
                     (int)224,                             // tensor_dim0_stride = 224
                     (int)(50176u << 16),                  // tensor_dim1_stride = 224*224 (lo16)
                     0 };
        i32x4 g2 = { 192, 0, 50176, 0 };                   // tensor_dim2, -, dim2_stride, -
        i32x4 g3 = { 0, 0, 0, 0 };
        i32x8 g4 = { 0, 0, 0, 0, 0, 0, 0, 0 };             // unused trailing group (6-arg form)
        __builtin_amdgcn_tensor_load_to_lds(g0, g1, g2, g3, g4, 0);
        __builtin_amdgcn_s_wait_tensorcnt(0);
    }
    __syncthreads();

    // ---- phase 1: transpose [c][m] f32 -> [m][c] bf16, zero-pad tokens 49..63 ----
    for (int i = tid; i < MPAD * C_IN; i += 256) {
        int m = i / C_IN;
        int c = i - m * C_IN;
        float v = (m < NTOK) ? sm.u.xwf[c * NTOK + m] : 0.0f;
        sm.xwbf[m * KSTR + c] = f2bf(v);
    }
    __syncthreads();

    // ---- phase 2: GEMM1  qk[64x384] = xw[64x192] @ W^T, +bias, q*scale, -> bf16 LDS ----
    for (int nt = wid; nt < 24; nt += 8) {
        const int n_g = nt * 16 + ln;
        Frag16 B6[6];
        const unsigned short* wrow = Wbf + n_g * C_IN;
        #pragma unroll
        for (int kt = 0; kt < 6; kt++) {
            const uv4* p = (const uv4*)(wrow + kt * 32 + 16 * lh);
            B6[kt].q[0] = p[0];
            B6[kt].q[1] = p[1];
        }
        const float bias = bqk[n_g];

        // wave-uniform Q/K selection (branchless element stores)
        const bool isq = __builtin_amdgcn_readfirstlane(nt < 12 ? 1 : 0) != 0;
        unsigned short* dstbase = isq ? (sm.qbf + n_g) : (sm.kbf + (n_g - C_IN));
        const float mul = isq ? QSCALE : 1.0f;

        for (int mt = 0; mt < 4; mt++) {
            v8f acc = {};
            // software-pipelined A-fragment double buffer
            Frag16 A[2];
            {
                const unsigned short* a0 = sm.xwbf + (mt * 16 + ln) * KSTR + 16 * lh;
                A[0].q[0] = *(const uv4*)(a0);
                A[0].q[1] = *(const uv4*)(a0 + 8);
            }
            #pragma unroll
            for (int kt = 0; kt < 6; kt++) {
                if (kt < 5) {
                    const unsigned short* an =
                        sm.xwbf + (mt * 16 + ln) * KSTR + (kt + 1) * 32 + 16 * lh;
                    A[(kt + 1) & 1].q[0] = *(const uv4*)(an);
                    A[(kt + 1) & 1].q[1] = *(const uv4*)(an + 8);
                }
                acc = __builtin_amdgcn_wmma_f32_16x16x32_bf16(
                        false, A[kt & 1].v, false, B6[kt].v, (short)0, acc, false, false);
            }
            #pragma unroll
            for (int r = 0; r < 8; r++) {
                int m = mt * 16 + r + 8 * lh;
                dstbase[m * KSTR] = f2bf((acc[r] + bias) * mul);
            }
        }
    }
    __syncthreads();

    // ---- phase 3: GEMM2  scores[h][n][m] = q[n,h,:] . k[m,h,:]  (K = 32, one WMMA) ----
    for (int t = wid; t < 96; t += 8) {
        int h   = t >> 4;
        int r2  = t & 15;
        int qt  = r2 >> 2;      // query-row tile
        int ktl = r2 & 3;       // key-col tile
        Frag16 A, B;
        const unsigned short* qa = sm.qbf + (qt * 16 + ln) * KSTR + h * HD + 16 * lh;
        A.q[0] = *(const uv4*)(qa);
        A.q[1] = *(const uv4*)(qa + 8);
        const unsigned short* kb = sm.kbf + (ktl * 16 + ln) * KSTR + h * HD + 16 * lh;
        B.q[0] = *(const uv4*)(kb);
        B.q[1] = *(const uv4*)(kb + 8);
        v8f s = {};
        s = __builtin_amdgcn_wmma_f32_16x16x32_bf16(
                false, A.v, false, B.v, (short)0, s, false, false);
        int m_g = ktl * 16 + ln;
        if (m_g < NTOK) {
            #pragma unroll
            for (int r = 0; r < 8; r++) {
                int n = qt * 16 + r + 8 * lh;
                if (n < NTOK) sm.u.scores[(h * NTOK + n) * SSTR + m_g] = s[r];
            }
        }
    }
    __syncthreads();

    // ---- phase 4: per-row max & sum(exp) ----
    for (int i = tid; i < NH * NTOK; i += 256) {
        const float* row = sm.u.scores + i * SSTR;
        float mx = -3.4e38f;
        #pragma unroll 7
        for (int j = 0; j < NTOK; j++) mx = fmaxf(mx, row[j]);
        float s = 0.0f;
        #pragma unroll 7
        for (int j = 0; j < NTOK; j++) s += __expf(row[j] - mx);
        sm.rowmax[i]    = mx;
        sm.rowsuminv[i] = 1.0f / s;
    }
    __syncthreads();

    // ---- phase 5: coalesced normalized writeback ----
    const size_t obase = (size_t)win * (NH * NTOK * NTOK);
    for (int i = tid; i < NH * NTOK * NTOK; i += 256) {
        int h = i / (NTOK * NTOK);
        int r = i - h * (NTOK * NTOK);
        int n = r / NTOK;
        int m = r - n * NTOK;
        int row = h * NTOK + n;
        out[obase + i] = __expf(sm.u.scores[row * SSTR + m] - sm.rowmax[row]) * sm.rowsuminv[row];
    }
}

// ---------- host ----------
extern "C" void kernel_launch(void* const* d_in, const int* in_sizes, int n_in,
                              void* d_out, int out_size, void* d_ws, size_t ws_size,
                              hipStream_t stream) {
    const float* x    = (const float*)d_in[0];   // (8,192,224,224)
    const float* W_qk = (const float*)d_in[1];   // (384,192)
    const float* b_qk = (const float*)d_in[2];   // (384,)
    float* out        = (float*)d_out;           // (8192,6,49,49)

    unsigned short* Wbf = (unsigned short*)d_ws; // 384*192 bf16 = 144 KB

    const int nw = C2 * C_IN;
    wcvt_kernel<<<(nw + 255) / 256, 256, 0, stream>>>(W_qk, Wbf, nw);
    win_attn_kernel<<<NWIN, 256, 0, stream>>>(x, Wbf, b_qk, out);
}
